// LinearAttention_25151328485889
// MI455X (gfx1250) — matrix operand order repaired and checked by reading, not compiled, not run on hardware
//
#include <hip/hip_runtime.h>

// ---------------------------------------------------------------------------
// Linear attention with 2nd-order Taylor feature map, reduced algebraically:
//   A_nm = 1 + (q_n.k_m)/4 + (q_n.k_m)^2/32   (d=16)
//   y    = tril(A) @ V ;  z = tril(A) @ 1 ;  y *= 1/(z+eps)
// CDNA5 (gfx1250, wave32): bf16 WMMA 16x16x32 for GEMMs + A@V,
// f32 WMMA 16x16x4 for S = Q K^T (precision before squaring).
// Pipelined fragment loads; double-buffered LDS staging in attention with
// GLOBAL_LOAD_ASYNC_TO_LDS (ASYNCcnt) for the K chunks.
// ---------------------------------------------------------------------------

#define DEV __device__ __forceinline__

typedef __attribute__((ext_vector_type(8)))  float  v8f;
typedef __attribute__((ext_vector_type(2)))  float  v2f;
typedef __attribute__((ext_vector_type(16))) __bf16 v16bf;

union BF16Frag { v16bf v; unsigned int u[8]; };

static constexpr int Bn = 2;
static constexpr int Ln = 2048;
static constexpr int Dm = 1024;
static constexpr int Hn = 16;
static constexpr int Fd = 16;   // feature dim per head (q,k)
static constexpr int Ed = 64;   // head dim (v)

#if defined(__has_builtin)
#if __has_builtin(__builtin_amdgcn_global_load_async_to_lds_b64)
#define ATHENA_ASYNC_LDS 1
#endif
#endif

#if defined(ATHENA_ASYNC_LDS)
typedef int v2i_vs __attribute__((vector_size(8)));
typedef __attribute__((address_space(1))) v2i_vs* gasync64_t;
typedef __attribute__((address_space(3))) v2i_vs* lasync64_t;
#endif

DEV unsigned short f2bf(float f) {
  unsigned int u = __builtin_bit_cast(unsigned int, f);
  u += 0x7FFFu + ((u >> 16) & 1u);           // round-to-nearest-even
  return (unsigned short)(u >> 16);
}

DEV v8f vzero8() {
  v8f v;
#pragma unroll
  for (int i = 0; i < 8; ++i) v[i] = 0.0f;
  return v;
}

DEV void wait_async0() {
#if defined(__has_builtin) && __has_builtin(__builtin_amdgcn_s_wait_asynccnt)
  __builtin_amdgcn_s_wait_asynccnt(0);
#else
  asm volatile("s_wait_asynccnt 0" ::: "memory");
#endif
}

// ---------------------------------------------------------------------------
// One-shot f32 -> bf16 conversion of all five tensors (contiguous dest).
// ---------------------------------------------------------------------------
__global__ void cvt5_f32_to_bf16(const float* __restrict__ s0,
                                 const float* __restrict__ s1,
                                 const float* __restrict__ s2,
                                 const float* __restrict__ s3,
                                 const float* __restrict__ s4,
                                 unsigned short* __restrict__ out) {
  const int N0 = Bn * Ln * Dm;          // hidden states
  const int N1 = 256 * Dm;              // Wq
  const int N2 = 256 * Dm;              // Wk
  const int N3 = Dm * Dm;               // Wv
  const int N4 = Dm * Dm;               // Wo
  const int total = N0 + N1 + N2 + N3 + N4;
  for (int i = blockIdx.x * blockDim.x + threadIdx.x; i < total;
       i += gridDim.x * blockDim.x) {
    int j = i;
    float v;
    if (j < N0) v = s0[j];
    else { j -= N0;
      if (j < N1) v = s1[j];
      else { j -= N1;
        if (j < N2) v = s2[j];
        else { j -= N2;
          if (j < N3) v = s3[j];
          else v = s4[j - N3];
        }
      }
    }
    out[i] = f2bf(v);
  }
}

// ---------------------------------------------------------------------------
// C[m,n] = sum_k A[m,k] * W[n,k]   (A:[M][K] bf16 row-major, W:[N][K] bf16)
// WG = 256 threads = 8 waves; per-WG tile 128(M) x 64(N); wave owns 16 rows.
// Double-buffered fragment loads (prefetch k+32 while computing k).
// gridSplit>0: blocks >= gridSplit use (W2,out2) instead (fused q+k GEMMs).
// mode 0: store f32 q/k layout [(b*16+h)][l][16]   (n = h*16+f, m = b*L+l)
// mode 1: store f32 v   layout [(b*16+h)][l][64]   (n = h*64+e)
// mode 2: store f32 row-major [m][n]
// ---------------------------------------------------------------------------
__global__ __launch_bounds__(256) void gemm_bf16_tn(
    const unsigned short* __restrict__ A, const unsigned short* __restrict__ W,
    float* __restrict__ out, int M, int N, int K, int mode, int gridSplit,
    const unsigned short* __restrict__ W2, float* __restrict__ out2) {
  int bid = blockIdx.x;
  if (gridSplit > 0 && bid >= gridSplit) {
    W = W2;
    out = out2;
    bid -= gridSplit;
  }
  const int tilesN = N >> 6;
  const int tm = bid / tilesN;
  const int tn = bid % tilesN;
  const int wave = threadIdx.x >> 5;
  const int lane = threadIdx.x & 31;
  const int hb = lane >> 4;      // lane half (0/1)
  const int r  = lane & 15;
  const int m0 = tm * 128 + wave * 16;
  const int n0 = tn * 64;

  v8f acc[4];
#pragma unroll
  for (int c = 0; c < 4; ++c) acc[c] = vzero8();

  // A fragment: lane(row=r) holds K halves [16*hb .. 16*hb+15]
  const unsigned short* arow = A + (size_t)(m0 + r) * K + 16 * hb;
  // B fragment: lane(col=n0+c*16+r) holds K halves [16*hb .. 16*hb+15]
  const unsigned short* wrow[4];
#pragma unroll
  for (int c = 0; c < 4; ++c)
    wrow[c] = W + (size_t)(n0 + c * 16 + r) * K + 16 * hb;

  auto ldA = [&](int kb, BF16Frag& f) {
    uint4 a0 = *(const uint4*)(arow + kb);
    uint4 a1 = *(const uint4*)(arow + kb + 8);
    f.u[0] = a0.x; f.u[1] = a0.y; f.u[2] = a0.z; f.u[3] = a0.w;
    f.u[4] = a1.x; f.u[5] = a1.y; f.u[6] = a1.z; f.u[7] = a1.w;
  };
  auto ldB = [&](int kb, int c, BF16Frag& f) {
    const uint4* pb = (const uint4*)(wrow[c] + kb);
    uint4 b0 = pb[0], b1 = pb[1];
    f.u[0] = b0.x; f.u[1] = b0.y; f.u[2] = b0.z; f.u[3] = b0.w;
    f.u[4] = b1.x; f.u[5] = b1.y; f.u[6] = b1.z; f.u[7] = b1.w;
  };

  BF16Frag af, bf[4];
  ldA(0, af);
#pragma unroll
  for (int c = 0; c < 4; ++c) ldB(0, c, bf[c]);

  for (int kb = 32; kb < K; kb += 32) {
    BF16Frag an, bn[4];
    ldA(kb, an);                    // prefetch next K-step
#pragma unroll
    for (int c = 0; c < 4; ++c) ldB(kb, c, bn[c]);
#pragma unroll
    for (int c = 0; c < 4; ++c)     // compute current K-step
      acc[c] = __builtin_amdgcn_wmma_f32_16x16x32_bf16(
          false, af.v, false, bf[c].v, (short)0, acc[c], false, false);
    af = an;
#pragma unroll
    for (int c = 0; c < 4; ++c) bf[c] = bn[c];
  }
#pragma unroll
  for (int c = 0; c < 4; ++c)
    acc[c] = __builtin_amdgcn_wmma_f32_16x16x32_bf16(
        false, af.v, false, bf[c].v, (short)0, acc[c], false, false);

  // Epilogue. C layout: element (M = rr + 8*hb, N = r) lives in acc[c][rr].
#pragma unroll
  for (int c = 0; c < 4; ++c) {
#pragma unroll
    for (int rr = 0; rr < 8; ++rr) {
      const int m = m0 + rr + 8 * hb;
      const int n = n0 + c * 16 + r;
      size_t idx;
      if (mode == 0) {
        const int b = m >> 11, lp = m & (Ln - 1);
        const int h = n >> 4, f = n & 15;
        idx = ((size_t)(b * Hn + h) * Ln + lp) * Fd + f;
      } else if (mode == 1) {
        const int b = m >> 11, lp = m & (Ln - 1);
        const int h = n >> 6, e = n & 63;
        idx = ((size_t)(b * Hn + h) * Ln + lp) * Ed + e;
      } else {
        idx = (size_t)m * N + n;
      }
      out[idx] = acc[c][rr];
    }
  }
}

// ---------------------------------------------------------------------------
// Causal Taylor attention, one (b,h, 128-row block) per WG; 8 waves x 16 rows.
// Double-buffered K/V staging: next chunk's loads issue before this chunk's
// WMMA chain so memory latency hides under matrix ops.
// ---------------------------------------------------------------------------
__global__ __launch_bounds__(256) void taylor_attn(
    const float* __restrict__ Q, const float* __restrict__ Kf,
    const float* __restrict__ V, unsigned short* __restrict__ Y) {
  __shared__ __align__(16) float Ks[2][32 * 16];              // K chunk  [t][f]
  __shared__ __align__(16) unsigned short Vt[2][64 * 34];     // V^T bf16 [e][t]
  __shared__ __align__(16) unsigned short Ast[8][16 * 36];    // per-wave A strip
  __shared__ float Zb[8][16];

  const int bh = blockIdx.x >> 4;         // 0..31
  const int rb = blockIdx.x & 15;         // row block 0..15
  const int b  = bh >> 4;
  const int h  = bh & 15;
  const int wave = threadIdx.x >> 5;
  const int lane = threadIdx.x & 31;
  const int hb = lane >> 4;
  const int r  = lane & 15;

  const float* Qb = Q  + (size_t)bh * Ln * Fd;
  const float* Kb = Kf + (size_t)bh * Ln * Fd;
  const float* Vb = V  + (size_t)bh * Ln * Ed;

  const int row0 = rb * 128 + wave * 16;  // wave's first query row

  // ---- staging helpers ----
  auto stageK = [&](int c0, int buf) {
#if defined(ATHENA_ASYNC_LDS)
    gasync64_t gp = (gasync64_t)(Kb + (size_t)c0 * Fd);
    lasync64_t lp = (lasync64_t)(&Ks[buf][0]);
    __builtin_amdgcn_global_load_async_to_lds_b64(gp + threadIdx.x,
                                                  lp + threadIdx.x, 0, 0);
#else
    ((float2*)&Ks[buf][0])[threadIdx.x] =
        ((const float2*)(Kb + (size_t)c0 * Fd))[threadIdx.x];
#endif
  };
  const int t_v  = threadIdx.x >> 3;       // time index within chunk
  const int e0_v = (threadIdx.x & 7) * 8;  // first of 8 head dims
  auto loadV = [&](int c0, float4& a, float4& bb) {
    const float4* pv = (const float4*)(Vb + (size_t)(c0 + t_v) * Ed + e0_v);
    a = pv[0]; bb = pv[1];
  };
  auto storeV = [&](int buf, float4 a, float4 bb) {
    unsigned short* d = &Vt[buf][0] + t_v;
    d[(e0_v + 0) * 34] = f2bf(a.x);  d[(e0_v + 1) * 34] = f2bf(a.y);
    d[(e0_v + 2) * 34] = f2bf(a.z);  d[(e0_v + 3) * 34] = f2bf(a.w);
    d[(e0_v + 4) * 34] = f2bf(bb.x); d[(e0_v + 5) * 34] = f2bf(bb.y);
    d[(e0_v + 6) * 34] = f2bf(bb.z); d[(e0_v + 7) * 34] = f2bf(bb.w);
  };

  // Q fragments for v_wmma_f32_16x16x4_f32 (A 16x4, 4 K-steps cover f=16).
  v2f qa[4];
#pragma unroll
  for (int c = 0; c < 4; ++c) {
    const float* p = Qb + (size_t)(row0 + r) * Fd + 4 * c + 2 * hb;
    v2f t; t.x = p[0]; t.y = p[1];
    qa[c] = t;
  }

  v8f acc[4];
#pragma unroll
  for (int c = 0; c < 4; ++c) acc[c] = vzero8();
  v8f zacc = vzero8();

  BF16Frag ones;
#pragma unroll
  for (int j = 0; j < 8; ++j) ones.u[j] = 0x3F803F80u;  // pair of bf16(1.0)

  const int nchunks = rb * 4 + 4;  // chunks of 32 keys, up to end of row block

  // ---- prologue: stage chunk 0 into buffer 0 ----
  stageK(0, 0);
  {
    float4 a, bb;
    loadV(0, a, bb);
    storeV(0, a, bb);
  }
#if defined(ATHENA_ASYNC_LDS)
  wait_async0();
#endif
  __syncthreads();

  for (int jc = 0; jc < nchunks; ++jc) {
    const int c0 = jc * 32;
    const int cur = jc & 1, nxt = cur ^ 1;
    const bool pf = (jc + 1 < nchunks);

    // ---- issue next chunk's staging before this chunk's compute ----
    float4 pva, pvb;
    if (pf) {
      stageK(c0 + 32, nxt);
      loadV(c0 + 32, pva, pvb);
      if (jc + 2 < nchunks) {
        __builtin_prefetch(Kb + (size_t)(c0 + 64) * Fd + threadIdx.x * 8, 0, 1);
        __builtin_prefetch(Vb + (size_t)(c0 + 64) * Ed + threadIdx.x * 32, 0, 1);
      }
    }

    // ---- preload all K fragments (B 4x16 f32), then S-WMMA chain ----
    // B lane(col n=r) holds B[2*hb][n], B[2*hb+1][n] = Ks[n][4c+2hb ..]
    v2f kfr[2][4];
#pragma unroll
    for (int st = 0; st < 2; ++st)
#pragma unroll
      for (int c = 0; c < 4; ++c) {
        const float* kp = &Ks[cur][0] + (16 * st + r) * Fd + 4 * c + 2 * hb;
        v2f t; t.x = kp[0]; t.y = kp[1];
        kfr[st][c] = t;
      }

#pragma unroll
    for (int st = 0; st < 2; ++st) {
      v8f s = vzero8();
#pragma unroll
      for (int c = 0; c < 4; ++c)
        s = __builtin_amdgcn_wmma_f32_16x16x4_f32(
            false, qa[c], false, kfr[st][c], (short)0, s, false, false);

      const int colg = c0 + st * 16 + r;  // this lane's key column (C layout N)
      unsigned short* ap = &Ast[wave][0];
#pragma unroll
      for (int rr = 0; rr < 8; ++rr) {
        const int rowg = row0 + rr + 8 * hb;
        const float sv = s[rr];
        float a = 1.0f + sv * 0.25f + sv * sv * 0.03125f;
        if (colg > rowg) a = 0.0f;  // causal (tril keeps diagonal)
        ap[(rr + 8 * hb) * 36 + st * 16 + r] = f2bf(a);
      }
    }

    // Wave-private LDS round-trip (C-layout -> A-fragment re-layout).
    asm volatile("s_wait_dscnt 0" ::: "memory");

    BF16Frag af;
    {
      const unsigned int* base = (const unsigned int*)&Ast[wave][0];
      const int ro = r * 18 + 8 * hb;  // dword offset: row r, halves 2j + 16*hb
#pragma unroll
      for (int j = 0; j < 4; ++j) af.u[j] = base[ro + j];
#pragma unroll
      for (int j = 0; j < 4; ++j) af.u[4 + j] = base[ro + 4 + j];
    }

    // ---- preload all V fragments, then y/z WMMA chain ----
    BF16Frag vf[4];
#pragma unroll
    for (int c = 0; c < 4; ++c) {
      const unsigned int* vb =
          (const unsigned int*)&Vt[cur][0] + (c * 16 + r) * 17 + 8 * hb;
#pragma unroll
      for (int j = 0; j < 8; ++j) vf[c].u[j] = vb[j];
    }
#pragma unroll
    for (int c = 0; c < 4; ++c)
      acc[c] = __builtin_amdgcn_wmma_f32_16x16x32_bf16(
          false, af.v, false, vf[c].v, (short)0, acc[c], false, false);
    zacc = __builtin_amdgcn_wmma_f32_16x16x32_bf16(
        false, af.v, false, ones.v, (short)0, zacc, false, false);

    // ---- finish next chunk's V staging, close the pipeline stage ----
    if (pf) storeV(nxt, pva, pvb);
#if defined(ATHENA_ASYNC_LDS)
    wait_async0();
#endif
    __syncthreads();
  }

  // Broadcast z (column 0 of zacc lives in lanes with N==0) via LDS
  if (r == 0) {
#pragma unroll
    for (int rr = 0; rr < 8; ++rr) Zb[wave][rr + 8 * hb] = zacc[rr];
  }
  asm volatile("s_wait_dscnt 0" ::: "memory");

  float zi[8];
#pragma unroll
  for (int rr = 0; rr < 8; ++rr)
    zi[rr] = 1.0f / (Zb[wave][rr + 8 * hb] + 1e-12f);

  // Store y as bf16, layout [token][h*64 + e] (matches reference concat order)
#pragma unroll
  for (int c = 0; c < 4; ++c) {
#pragma unroll
    for (int rr = 0; rr < 8; ++rr) {
      const int rowg = row0 + rr + 8 * hb;
      const int token = b * Ln + rowg;
      const int col = h * Ed + c * 16 + r;
      Y[(size_t)token * Dm + col] = f2bf(acc[c][rr] * zi[rr]);
    }
  }
}

// ---------------------------------------------------------------------------
extern "C" void kernel_launch(void* const* d_in, const int* in_sizes, int n_in,
                              void* d_out, int out_size, void* d_ws,
                              size_t ws_size, hipStream_t stream) {
  (void)in_sizes; (void)n_in; (void)out_size; (void)ws_size;
  const float* hs = (const float*)d_in[0];
  const float* Wq = (const float*)d_in[1];
  const float* Wk = (const float*)d_in[2];
  const float* Wv = (const float*)d_in[3];
  const float* Wo = (const float*)d_in[4];
  float* out = (float*)d_out;

  const int M = Bn * Ln;  // 4096 tokens

  char* ws = (char*)d_ws;
  size_t off = 0;
  auto alloc = [&](size_t bytes) {
    void* p = ws + off;
    off += (bytes + 255) & ~(size_t)255;
    return p;
  };
  // NOTE: the five bf16 regions are contiguous (all sizes multiple of 256B) so
  // cvt5 writes them as one flat destination.
  unsigned short* hsb = (unsigned short*)alloc((size_t)M * Dm * 2);
  unsigned short* wqb = (unsigned short*)alloc((size_t)256 * Dm * 2);
  unsigned short* wkb = (unsigned short*)alloc((size_t)256 * Dm * 2);
  unsigned short* wvb = (unsigned short*)alloc((size_t)Dm * Dm * 2);
  unsigned short* wob = (unsigned short*)alloc((size_t)Dm * Dm * 2);
  float* q_ws = (float*)alloc((size_t)Bn * Hn * Ln * Fd * 4);
  float* k_ws = (float*)alloc((size_t)Bn * Hn * Ln * Fd * 4);
  float* v_ws = (float*)alloc((size_t)Bn * Hn * Ln * Ed * 4);
  unsigned short* yb = (unsigned short*)alloc((size_t)M * Dm * 2);

  // 1) convert everything to bf16 in one launch
  cvt5_f32_to_bf16<<<4096, 256, 0, stream>>>(hs, Wq, Wk, Wv, Wo, hsb);

  // 2) fused q+k projections (one launch, split grid), then v projection
  gemm_bf16_tn<<<2 * (M / 128) * (256 / 64), 256, 0, stream>>>(
      hsb, wqb, q_ws, M, 256, Dm, 0, (M / 128) * (256 / 64), wkb, k_ws);
  gemm_bf16_tn<<<(M / 128) * (Dm / 64), 256, 0, stream>>>(
      hsb, wvb, v_ws, M, Dm, Dm, 1, 0, nullptr, nullptr);

  // 3) causal Taylor attention: 32 (b,h) pairs x 16 row blocks
  taylor_attn<<<Bn * Hn * (Ln / 128), 256, 0, stream>>>(q_ws, k_ws, v_ws, yb);

  // 4) output projection
  gemm_bf16_tn<<<(M / 128) * (Dm / 64), 256, 0, stream>>>(
      yb, wob, out, M, Dm, Dm, 2, 0, nullptr, nullptr);
}
